// ADEA_52759378264204
// MI455X (gfx1250) — matrix operand order repaired and checked by reading, not compiled, not run on hardware
//
#include <hip/hip_runtime.h>
#include <hip/hip_bf16.h>
#include <math.h>

// ---------------------------------------------------------------------------
// ADEA graph-attention forward for MI455X (gfx1250, wave32).
// Memory-bound (~2-3 GB traffic @ 23.3 TB/s); WMMA f16 16x16x32 used for all
// per-edge attention-score matvecs (16-edge tiles, K chunks of 32).
// ---------------------------------------------------------------------------

typedef __attribute__((ext_vector_type(16))) _Float16 v16h;
typedef __attribute__((ext_vector_type(8)))  float    v8f;

__device__ __forceinline__ float leaky(float v) { return v > 0.f ? v : 0.3f * v; }

__device__ __forceinline__ float waveMax(float v) {
#pragma unroll
  for (int o = 16; o > 0; o >>= 1) v = fmaxf(v, __shfl_xor(v, o, 32));
  return v;
}
__device__ __forceinline__ float waveSum(float v) {
#pragma unroll
  for (int o = 16; o > 0; o >>= 1) v += __shfl_xor(v, o, 32);
  return v;
}

__device__ __forceinline__ unsigned floatFlip(float f) {
  unsigned u = __float_as_uint(f);
  return (u & 0x80000000u) ? ~u : (u | 0x80000000u);
}
__device__ __forceinline__ float floatUnflip(unsigned u) {
  u = (u & 0x80000000u) ? (u & 0x7FFFFFFFu) : ~u;
  return __uint_as_float(u);
}

// ---------------------------------------------------------------------------
// Row-pointer build: edges are sorted by row, so starts[n] = lower_bound(row,n)
// ---------------------------------------------------------------------------
__global__ void lb_kernel(const int* __restrict__ pairs, int count,
                          int* __restrict__ starts, int N_) {
  int i = blockIdx.x * blockDim.x + threadIdx.x;
  if (i > N_) return;
  int lo = 0, hi = count;
  while (lo < hi) {
    int mid = (lo + hi) >> 1;
    if (pairs[(long)mid * 2] < i) lo = mid + 1; else hi = mid;
  }
  starts[i] = lo;
}

// rels_mean[e] = 0.5*(rel_emb[er_rel[2e]] + rel_emb[er_rel[2e+1]])
__global__ void rels_mean_kernel(const float* __restrict__ rel_emb,
                                 const int* __restrict__ er_rel,
                                 float* __restrict__ rm, int E_) {
  long t = blockIdx.x * (long)blockDim.x + threadIdx.x;
  long total = (long)E_ * 16;
  if (t >= total) return;
  long e = t >> 4; int q = (int)(t & 15) * 4;
  int i0 = er_rel[2 * e], i1 = er_rel[2 * e + 1];
  const float4 a = *(const float4*)(rel_emb + (long)i0 * 64 + q);
  const float4 b = *(const float4*)(rel_emb + (long)i1 * 64 + q);
  float4 r; r.x = (a.x + b.x) * 0.5f; r.y = (a.y + b.y) * 0.5f;
  r.z = (a.z + b.z) * 0.5f; r.w = (a.w + b.w) * 0.5f;
  *(float4*)(rm + e * 64 + q) = r;
}

// ---------------------------------------------------------------------------
// Concept edge scores: a_e = leaky([ent_emb[row] | emb[idx]] . w + b), K=192.
// One wave handles 16 edges; 6 WMMA f16 K-chunks.
// ---------------------------------------------------------------------------
__global__ void concept_score_kernel(const float* __restrict__ ent_emb,
                                     const float* __restrict__ emb,
                                     const int* __restrict__ pair, // (M,2) row,idx
                                     const float* __restrict__ w,  // 192
                                     const float* __restrict__ bp, // 1
                                     float* __restrict__ scores, int M) {
  long wave = (blockIdx.x * (long)blockDim.x + threadIdx.x) >> 5;
  int lane = threadIdx.x & 31;
  long eBase = wave * 16;
  if (eBase >= M) return;
  int m = lane & 15;
  long e = eBase + m; if (e >= M) e = M - 1;
  int row = pair[2 * e], idx = pair[2 * e + 1];
  const float* fr = ent_emb + (long)row * 128;
  const float* fi = emb + (long)idx * 64;
  int hi = lane >> 4;
  v8f c = {};
#pragma unroll
  for (int ch = 0; ch < 6; ++ch) {
    int k0 = ch * 32 + hi * 8;
    v16h a;
#pragma unroll
    for (int s = 0; s < 2; ++s) {
      int kk = k0 + s * 16;
      const float* src = (kk < 128) ? (fr + kk) : (fi + (kk - 128));
#pragma unroll
      for (int j = 0; j < 8; ++j) a[s * 8 + j] = (_Float16)src[j];
    }
    v16h bm; int kb = ch * 32 + hi * 8;
#pragma unroll
    for (int j = 0; j < 16; ++j) bm[j] = (_Float16)w[kb + (j < 8 ? j : j + 8)];
    c = __builtin_amdgcn_wmma_f32_16x16x32_f16(false, a, false, bm, (short)0, c,
                                               false, false);
  }
  if ((lane & 15) == 0) {
    float bb = bp[0];
#pragma unroll
    for (int j = 0; j < 8; ++j) {
      long oe = eBase + hi * 8 + j;
      if (oe < M) scores[oe] = leaky(c[j] + bb);
    }
  }
}

// Segment softmax + weighted sum + relu:  out[n] = relu(sum p_e * emb[idx_e])
__global__ void concept_agg_kernel(const float* __restrict__ scores,
                                   const int* __restrict__ pair,
                                   const float* __restrict__ emb,
                                   const int* __restrict__ starts, int mul,
                                   float* __restrict__ out, int N_) {
  long wave = (blockIdx.x * (long)blockDim.x + threadIdx.x) >> 5;
  int lane = threadIdx.x & 31;
  if (wave >= N_) return;
  int s = starts[wave] * mul, e = starts[wave + 1] * mul;
  float m = -3.4e38f;
  for (int i = s + lane; i < e; i += 32) m = fmaxf(m, scores[i]);
  m = waveMax(m);
  float z = 0.f;
  for (int i = s + lane; i < e; i += 32) z += __expf(scores[i] - m);
  z = waveSum(z);
  float a0 = 0.f, a1 = 0.f;
  if (e > s) {
    float inv = 1.f / z;
    for (int i = s; i < e; ++i) {
      float pe = __expf(scores[i] - m) * inv;
      const float* src = emb + (long)pair[2 * i + 1] * 64;
      a0 += pe * src[lane];
      a1 += pe * src[lane + 32];
    }
  }
  out[wave * 64 + lane]      = fmaxf(a0, 0.f);
  out[wave * 64 + lane + 32] = fmaxf(a1, 0.f);
}

// x0[n] = mean_{e in row n} ent_emb[col_e]
__global__ void mean_agg_kernel(const float* __restrict__ ent_emb,
                                const int* __restrict__ ee,
                                const int* __restrict__ starts,
                                float* __restrict__ x0, int N_) {
  long wave = (blockIdx.x * (long)blockDim.x + threadIdx.x) >> 5;
  int lane = threadIdx.x & 31;
  if (wave >= N_) return;
  int s = starts[wave], e = starts[wave + 1];
  float acc[4] = {0.f, 0.f, 0.f, 0.f};
  for (int i = s; i < e; ++i) {
    const float* src = ent_emb + (long)ee[2 * i + 1] * 128;
#pragma unroll
    for (int k = 0; k < 4; ++k) acc[k] += src[lane + 32 * k];
  }
  float inv = 1.f / fmaxf((float)(e - s), 1.f);
#pragma unroll
  for (int k = 0; k < 4; ++k) x0[wave * 128 + lane + 32 * k] = acc[k] * inv;
}

// ---------------------------------------------------------------------------
// con_attn scores for all 4 (layer,head) columns in ONE pass over con_feat
// (K=320, assembled on the fly; never materialized). 10 WMMA chunks / 16 edges.
// ---------------------------------------------------------------------------
__global__ void ca_score_kernel(const float* __restrict__ crel,
                                const float* __restrict__ cattr,
                                const float* __restrict__ rm,
                                const int* __restrict__ ee,
                                const float* __restrict__ caw, // (4,320)
                                const float* __restrict__ cab, // (4)
                                float* __restrict__ ca, int E_) {
  long wave = (blockIdx.x * (long)blockDim.x + threadIdx.x) >> 5;
  int lane = threadIdx.x & 31;
  long eBase = wave * 16;
  if (eBase >= E_) return;
  int m = lane & 15;
  long e = eBase + m; if (e >= E_) e = E_ - 1;
  int row = ee[2 * e], col = ee[2 * e + 1];
  const float* p0 = crel  + (long)row * 64;   // K   0..63
  const float* p1 = cattr + (long)row * 64;   // K  64..127
  const float* p2 = rm    + e * 64;           // K 128..191
  const float* p3 = crel  + (long)col * 64;   // K 192..255
  const float* p4 = cattr + (long)col * 64;   // K 256..319
  int hi = lane >> 4;
  int n = lane & 15;
  v8f c = {};
#pragma unroll
  for (int ch = 0; ch < 10; ++ch) {
    int k0 = ch * 32 + hi * 8;
    v16h a;
#pragma unroll
    for (int s = 0; s < 2; ++s) {
      int kk = k0 + s * 16;
      const float* src;
      if (kk < 64)       src = p0 + kk;
      else if (kk < 128) src = p1 + (kk - 64);
      else if (kk < 192) src = p2 + (kk - 128);
      else if (kk < 256) src = p3 + (kk - 192);
      else               src = p4 + (kk - 256);
#pragma unroll
      for (int j = 0; j < 8; ++j) a[s * 8 + j] = (_Float16)src[j];
    }
    v16h bm; int kb = ch * 32 + hi * 8;
#pragma unroll
    for (int j = 0; j < 16; ++j)
      bm[j] = (n < 4) ? (_Float16)caw[n * 320 + kb + (j < 8 ? j : j + 8)] : (_Float16)0.0f;
    c = __builtin_amdgcn_wmma_f32_16x16x32_f16(false, a, false, bm, (short)0, c,
                                               false, false);
  }
  if (n < 4) {
    float bb = cab[n];
#pragma unroll
    for (int j = 0; j < 8; ++j) {
      long oe = eBase + hi * 8 + j;
      if (oe < E_) ca[oe * 4 + n] = leaky(c[j] + bb);
    }
  }
}

// ent_attn scores for one (l,h):  v_e = leaky(ef . w + b) * ca[e][lh], K=192
__global__ void ea_score_kernel(const float* __restrict__ x_prev,
                                const float* __restrict__ rm,
                                const int* __restrict__ ee,
                                const float* __restrict__ wv,  // 192
                                const float* __restrict__ bp,  // 1
                                const float* __restrict__ ca, int lh,
                                float* __restrict__ out, int E_, int hOff) {
  long wave = (blockIdx.x * (long)blockDim.x + threadIdx.x) >> 5;
  int lane = threadIdx.x & 31;
  long eBase = wave * 16;
  if (eBase >= E_) return;
  int m = lane & 15;
  long e = eBase + m; if (e >= E_) e = E_ - 1;
  int row = ee[2 * e], col = ee[2 * e + 1];
  const float* fr = x_prev + (long)row * 128 + hOff; // relu applied
  const float* fm = rm + e * 64;
  const float* fc = x_prev + (long)col * 128 + hOff; // relu applied
  int hi = lane >> 4;
  v8f c = {};
#pragma unroll
  for (int ch = 0; ch < 6; ++ch) {
    int k0 = ch * 32 + hi * 8;
    v16h a;
#pragma unroll
    for (int s = 0; s < 2; ++s) {
      int kk = k0 + s * 16;
      if (kk < 64) {
#pragma unroll
        for (int j = 0; j < 8; ++j) a[s * 8 + j] = (_Float16)fmaxf(fr[kk + j], 0.f);
      } else if (kk < 128) {
#pragma unroll
        for (int j = 0; j < 8; ++j) a[s * 8 + j] = (_Float16)fm[kk - 64 + j];
      } else {
#pragma unroll
        for (int j = 0; j < 8; ++j) a[s * 8 + j] = (_Float16)fmaxf(fc[kk - 128 + j], 0.f);
      }
    }
    v16h bm; int kb = ch * 32 + hi * 8;
#pragma unroll
    for (int j = 0; j < 16; ++j) bm[j] = (_Float16)wv[kb + (j < 8 ? j : j + 8)];
    c = __builtin_amdgcn_wmma_f32_16x16x32_f16(false, a, false, bm, (short)0, c,
                                               false, false);
  }
  if ((lane & 15) == 0) {
    float bb = bp[0];
#pragma unroll
    for (int j = 0; j < 8; ++j) {
      long oe = eBase + hi * 8 + j;
      if (oe < E_) out[oe] = leaky(c[j] + bb) * ca[oe * 4 + lh];
    }
  }
}

// --------------------- global softmax over all E edges ----------------------
__global__ void init_slots_kernel(unsigned* slotM, float* slotS) {
  if (threadIdx.x == 0 && blockIdx.x == 0) { *slotM = 0u; *slotS = 0.f; }
}
__global__ void gmax_kernel(const float* __restrict__ v, int n, unsigned* slotM) {
  float m = -3.4e38f;
  for (long i = blockIdx.x * (long)blockDim.x + threadIdx.x; i < n;
       i += (long)gridDim.x * blockDim.x) m = fmaxf(m, v[i]);
  m = waveMax(m);
  __shared__ float sm[8];
  if ((threadIdx.x & 31) == 0) sm[threadIdx.x >> 5] = m;
  __syncthreads();
  if (threadIdx.x == 0) {
    float t = sm[0];
    for (int k = 1; k < (int)(blockDim.x >> 5); ++k) t = fmaxf(t, sm[k]);
    atomicMax(slotM, floatFlip(t));
  }
}
__global__ void gsum_kernel(const float* __restrict__ v, int n,
                            const unsigned* slotM, float* slotS) {
  float M = floatUnflip(*slotM);
  float s = 0.f;
  for (long i = blockIdx.x * (long)blockDim.x + threadIdx.x; i < n;
       i += (long)gridDim.x * blockDim.x) s += __expf(v[i] - M);
  s = waveSum(s);
  __shared__ float sm[8];
  if ((threadIdx.x & 31) == 0) sm[threadIdx.x >> 5] = s;
  __syncthreads();
  if (threadIdx.x == 0) {
    float t = 0.f;
    for (int k = 0; k < (int)(blockDim.x >> 5); ++k) t += sm[k];
    atomicAdd(slotS, t);
  }
}
__global__ void gnorm_kernel(float* __restrict__ v, int n,
                             const unsigned* slotM, const float* slotS) {
  long i = blockIdx.x * (long)blockDim.x + threadIdx.x;
  if (i >= n) return;
  float M = floatUnflip(*slotM);
  float inv = 1.f / *slotS;
  v[i] = __expf(v[i] - M) * inv;
}

// Segment softmax of p over rows + weighted sum of relu(x_prev[col]) + tanh
__global__ void seg_agg_kernel(const float* __restrict__ p,
                               const int* __restrict__ ee,
                               const int* __restrict__ starts,
                               const float* __restrict__ x_prev,
                               float* __restrict__ x_next, int hOff, int N_) {
  long wave = (blockIdx.x * (long)blockDim.x + threadIdx.x) >> 5;
  int lane = threadIdx.x & 31;
  if (wave >= N_) return;
  int s = starts[wave], e = starts[wave + 1];
  float m = -3.4e38f;
  for (int i = s + lane; i < e; i += 32) m = fmaxf(m, p[i]);
  m = waveMax(m);
  float z = 0.f;
  for (int i = s + lane; i < e; i += 32) z += __expf(p[i] - m);
  z = waveSum(z);
  float a0 = 0.f, a1 = 0.f;
  if (e > s) {
    float inv = 1.f / z;
    for (int i = s; i < e; ++i) {
      float pe = __expf(p[i] - m) * inv;
      const float* nf = x_prev + (long)ee[2 * i + 1] * 128 + hOff;
      a0 += pe * fmaxf(nf[lane], 0.f);
      a1 += pe * fmaxf(nf[lane + 32], 0.f);
    }
  }
  x_next[wave * 128 + hOff + lane]      = tanhf(a0);
  x_next[wave * 128 + hOff + lane + 32] = tanhf(a1);
}

// d_out = [ent_out (N,384) = x1|x2|crel|cattr ; conc (N,128) = crel|cattr]
__global__ void pack_kernel(const float* __restrict__ x1,
                            const float* __restrict__ x2,
                            const float* __restrict__ crel,
                            const float* __restrict__ cattr,
                            float* __restrict__ out, int N_) {
  long i = blockIdx.x * (long)blockDim.x + threadIdx.x;
  long t1 = (long)N_ * 384;
  long total = t1 + (long)N_ * 128;
  if (i >= total) return;
  float v;
  if (i < t1) {
    long n = i / 384; int j = (int)(i % 384);
    v = (j < 128) ? x1[n * 128 + j]
      : (j < 256) ? x2[n * 128 + (j - 128)]
      : (j < 320) ? crel[n * 64 + (j - 256)]
                  : cattr[n * 64 + (j - 320)];
  } else {
    long k = i - t1; long n = k / 128; int j = (int)(k % 128);
    v = (j < 64) ? crel[n * 64 + j] : cattr[n * 64 + (j - 64)];
  }
  out[i] = v;
}

// ---------------------------------------------------------------------------
extern "C" void kernel_launch(void* const* d_in, const int* in_sizes, int n_in,
                              void* d_out, int out_size, void* d_ws, size_t ws_size,
                              hipStream_t stream) {
  (void)n_in; (void)out_size; (void)ws_size;
  const float* ent_emb  = (const float*)d_in[0];
  const float* rel_emb  = (const float*)d_in[1];
  const float* attr_emb = (const float*)d_in[2];
  const float* w_r = (const float*)d_in[3];
  const float* b_r = (const float*)d_in[4];
  const float* w_a = (const float*)d_in[5];
  const float* b_a = (const float*)d_in[6];
  const float* eaw = (const float*)d_in[7];   // (L,H,192,1)
  const float* eab = (const float*)d_in[8];   // (L,H,1)
  const float* caw = (const float*)d_in[9];   // (L,H,320,1)
  const float* cab = (const float*)d_in[10];  // (L,H,1)
  const int* ee       = (const int*)d_in[11];
  const int* er_rel   = (const int*)d_in[13];
  const int* er_index = (const int*)d_in[14];
  const int* ea_index = (const int*)d_in[15];

  const int N  = in_sizes[0] / 128;
  const int E  = in_sizes[11] / 2;
  const int E2 = in_sizes[13];
  const int EA = in_sizes[15] / 2;
  const int L = 2, H = 2;

  // scratch carve-out (~215 MB)
  char* wsb = (char*)d_ws;
  size_t off = 0;
  auto carve = [&](size_t bytes) -> void* {
    void* p = wsb + off;
    off = (off + bytes + 255) & ~(size_t)255;
    return p;
  };
  int*   ee_start = (int*)carve((size_t)(N + 1) * sizeof(int));
  int*   ea_start = (int*)carve((size_t)(N + 1) * sizeof(int));
  float* rm    = (float*)carve((size_t)E * 64 * sizeof(float));
  float* crel  = (float*)carve((size_t)N * 64 * sizeof(float));
  float* cattr = (float*)carve((size_t)N * 64 * sizeof(float));
  float* x0 = (float*)carve((size_t)N * 128 * sizeof(float));
  float* x1 = (float*)carve((size_t)N * 128 * sizeof(float));
  float* x2 = (float*)carve((size_t)N * 128 * sizeof(float));
  size_t scn = (size_t)(E2 > EA ? E2 : EA); if (scn < (size_t)E) scn = E;
  float* sc = (float*)carve(scn * sizeof(float));
  float* ca = (float*)carve((size_t)E * 4 * sizeof(float));
  unsigned* slotM = (unsigned*)carve(256);
  float* slotS = (float*)((char*)slotM + 4);

  const int T = 256;
  auto blocksThreads = [&](long n) { return (int)((n + T - 1) / T); };
  auto blocksWaves   = [&](long w) { return (int)((w + 7) / 8); };

  // Row pointers (edge lists sorted by row)
  lb_kernel<<<blocksThreads(N + 1), T, 0, stream>>>(ee, E, ee_start, N);
  lb_kernel<<<blocksThreads(N + 1), T, 0, stream>>>(ea_index, EA, ea_start, N);

  // rels_mean (E x 64)
  rels_mean_kernel<<<blocksThreads((long)E * 16), T, 0, stream>>>(rel_emb, er_rel, rm, E);

  // concept_rel
  concept_score_kernel<<<blocksWaves((E2 + 15) / 16), T, 0, stream>>>(
      ent_emb, rel_emb, er_index, w_r, b_r, sc, E2);
  concept_agg_kernel<<<blocksWaves(N), T, 0, stream>>>(
      sc, er_index, rel_emb, ee_start, 2, crel, N);

  // concept_attr
  concept_score_kernel<<<blocksWaves((EA + 15) / 16), T, 0, stream>>>(
      ent_emb, attr_emb, ea_index, w_a, b_a, sc, EA);
  concept_agg_kernel<<<blocksWaves(N), T, 0, stream>>>(
      sc, ea_index, attr_emb, ea_start, 1, cattr, N);

  // x0 = neighbor mean
  mean_agg_kernel<<<blocksWaves(N), T, 0, stream>>>(ent_emb, ee, ee_start, x0, N);

  // all 4 con_attn scores in one streaming pass over con_feat
  ca_score_kernel<<<blocksWaves((E + 15) / 16), T, 0, stream>>>(
      crel, cattr, rm, ee, caw, cab, ca, E);

  // GAT layers
  float* xp = x0;
  float* layers[2] = {x1, x2};
  for (int l = 0; l < L; ++l) {
    for (int h = 0; h < H; ++h) {
      int lh = l * H + h;
      ea_score_kernel<<<blocksWaves((E + 15) / 16), T, 0, stream>>>(
          xp, rm, ee, eaw + (size_t)lh * 192, eab + lh, ca, lh, sc, E, h * 64);
      init_slots_kernel<<<1, 32, 0, stream>>>(slotM, slotS);
      gmax_kernel<<<256, T, 0, stream>>>(sc, E, slotM);
      gsum_kernel<<<256, T, 0, stream>>>(sc, E, slotM, slotS);
      gnorm_kernel<<<blocksThreads(E), T, 0, stream>>>(sc, E, slotM, slotS);
      seg_agg_kernel<<<blocksWaves(N), T, 0, stream>>>(
          sc, ee, ee_start, xp, layers[l], h * 64, N);
    }
    xp = layers[l];
  }

  // Pack outputs
  pack_kernel<<<blocksThreads((long)N * 512), T, 0, stream>>>(
      x1, x2, crel, cattr, (float*)d_out, N);
}
